// MpnnGGNN_81157702025803
// MI455X (gfx1250) — matrix operand order repaired and checked by reading, not compiled, not run on hardware
//
#include <hip/hip_runtime.h>
#include <hip/hip_bf16.h>
#include <math.h>

// GGNN: B=64, N=256, IN=64, HID=MSG=256, TGT=128, 3 layers, 4 edge labels.
// All GEMMs on v_wmma_f32_16x16x32_f16 (f32 accumulate).
// - Hl stored transposed (HlT[l][b][m][i]) so aggregation B-fragments and the
//   k_msg D-store are per-lane contiguous (b128 traffic).
// - Edge labels packed once into a j-major byte map; mask A-fragments are two
//   u64 loads per K-step, expanded to 0/1 f16 in VALU, and each expansion is
//   reused across 4 output column tiles (4 accumulators per wave).

typedef __attribute__((ext_vector_type(16))) _Float16 v16h;
typedef __attribute__((ext_vector_type(8)))  _Float16 v8h;
typedef __attribute__((ext_vector_type(8)))  float    v8f;

__device__ __forceinline__ v8f wmma16(v16h a, v16h b, v8f c) {
  return __builtin_amdgcn_wmma_f32_16x16x32_f16(false, a, false, b, (short)0, c,
                                                false, false);
}

__device__ __forceinline__ float sigmoidf_(float x) {
  return 1.0f / (1.0f + __expf(-x));
}

// ---- WMMA fragment helpers (16x16x32 f16, wave32) -------------------------
// A (16x32, MxK), row-major source. Lane l: row = l&15, half = l>>4.
// VGPR v0..3 hold K = half*8 + {0..7}; v4..7 hold K = 16 + half*8 + {0..7}.
__device__ __forceinline__ v16h load_a_frag(const _Float16* p, int ld) {
  int l = threadIdx.x & 31;
  const _Float16* r0 = p + (l & 15) * ld + (l >> 4) * 8;
  v16h a;
#pragma unroll
  for (int i = 0; i < 8; ++i) a[i] = r0[i];
#pragma unroll
  for (int i = 0; i < 8; ++i) a[8 + i] = r0[16 + i];
  return a;
}

// B (32x16) sourced transposed: B[k][n] = p[n*ld + k]  (contiguous per lane).
// Lane l: col n = l&15, half = l>>4; b[0..7] hold K = half*8 + {0..7}; b[8..15] hold K = 16 + half*8 + {0..7}.
__device__ __forceinline__ v16h load_b_frag_t(const _Float16* p, int ld) {
  int l = threadIdx.x & 31;
  const _Float16* r0 = p + (l & 15) * ld + (l >> 4) * 8;
  v16h b;
#pragma unroll
  for (int i = 0; i < 8; ++i) b[i] = r0[i];
#pragma unroll
  for (int i = 0; i < 8; ++i) b[8 + i] = r0[16 + i];
  return b;
}

// C/D (16x16 f32): lane l holds col n=l&15; c[r] <-> row M = r + 8*(l>>4).
__device__ __forceinline__ void store_c_f16(_Float16* p, int ld, v8f c) {
  int l = threadIdx.x & 31, n = l & 15, half = l >> 4;
#pragma unroll
  for (int r = 0; r < 8; ++r) p[(r + 8 * half) * ld + n] = (_Float16)c[r];
}

// Contiguous per-lane store of the 8 accumulator rows (transposed layouts).
__device__ __forceinline__ void store_c_f16_contig(_Float16* p, v8f c) {
  v8h h;
#pragma unroll
  for (int r = 0; r < 8; ++r) h[r] = (_Float16)c[r];
  *(v8h*)p = h;
}

__device__ __forceinline__ void store_relu_f16(_Float16* p, int ld, v8f c,
                                               const float* bias, int colbase) {
  int l = threadIdx.x & 31, n = l & 15, half = l >> 4;
#pragma unroll
  for (int r = 0; r < 8; ++r) {
    float v = c[r] + bias[colbase + n];
    p[(r + 8 * half) * ld + n] = (_Float16)(v > 0.0f ? v : 0.0f);
  }
}

// Expand 16 packed label bytes into a 0/1 f16 A-fragment for one label.
__device__ __forceinline__ v16h mask_to_frag(unsigned long long m0,
                                             unsigned long long m1, int label) {
  v16h a;
#pragma unroll
  for (int i = 0; i < 8; ++i) {
    a[i]     = (_Float16)((int)((m0 >> (8 * i)) & 0xFF) == label ? 1.0f : 0.0f);
    a[8 + i] = (_Float16)((int)((m1 >> (8 * i)) & 0xFF) == label ? 1.0f : 0.0f);
  }
  return a;
}

// ---- kernels ---------------------------------------------------------------
__global__ void k_cvt(const float* __restrict__ s, _Float16* __restrict__ d, int n) {
  int i = blockIdx.x * 256 + threadIdx.x;
  if (i < n) d[i] = (_Float16)s[i];
}

// labT[b][j][i] = g[b,i,j] ? (u8)e[b,i,j] : 0   (j-major for contiguous K-runs)
__global__ void k_lab(const float* __restrict__ e, const float* __restrict__ g,
                      unsigned char* __restrict__ labT) {
  int idx = blockIdx.x * 256 + threadIdx.x;  // = b*65536 + j*256 + i
  int b = idx >> 16;
  int j = (idx >> 8) & 255;
  int i = idx & 255;
  int src = b * 65536 + i * 256 + j;
  labT[idx] = (g[src] != 0.0f) ? (unsigned char)(int)e[src] : (unsigned char)0;
}

// h_t = [h_in | zeros], plus f16 copy. grid=16384 (node), block=256 (channel)
__global__ void k_init(const float* __restrict__ hin, float* __restrict__ ht,
                       _Float16* __restrict__ h16) {
  int node = blockIdx.x, c = threadIdx.x;
  float v = (c < 64) ? hin[node * 64 + c] : 0.0f;
  ht[node * 256 + c] = v;
  h16[node * 256 + c] = (_Float16)v;
}

__global__ void k_valid(const float* __restrict__ hin, float* __restrict__ valid) {
  int node = blockIdx.x * 256 + threadIdx.x;
  float s = 0.0f;
  for (int c = 0; c < 64; ++c) s += fabsf(hin[node * 64 + c]);
  valid[node] = (s > 0.0f) ? 1.0f : 0.0f;
}

// HlT[l,b,m,i] = sum_h h16[(b,i),h] * A[l][m,h].  65536 wave-tiles.
__global__ void k_msg(const _Float16* __restrict__ h16,
                      const _Float16* __restrict__ A16,
                      _Float16* __restrict__ HlT) {
  int t = blockIdx.x * 4 + (threadIdx.x >> 5);
  int l = t >> 14;
  int r = t & 16383;
  int rt = r >> 4, ct = r & 15;
  const _Float16* ap = h16 + (rt * 16) * 256;
  const _Float16* bp = A16 + l * 65536 + (ct * 16) * 256;
  v8f acc = {};
#pragma unroll
  for (int k = 0; k < 8; ++k)
    acc = wmma16(load_a_frag(ap + k * 32, 256), load_b_frag_t(bp + k * 32, 256), acc);
  // transposed store: per lane contiguous 8 halves along i
  int lane = threadIdx.x & 31, n = lane & 15, half = lane >> 4;
  int b = rt >> 4;
  _Float16* dst = HlT + (((l * 64 + b) * 256) + ct * 16 + n) * 256
                      + (rt & 15) * 16 + 8 * half;
  store_c_f16_contig(dst, acc);
}

// m[b,j,m] = sum_l sum_i w_l[b,i,j] * Hl[l,b,i,m].  4096 waves, 4 mt tiles per
// wave: mask bytes + f16 expansion amortized over 4 WMMAs (128 WMMA per wave).
__global__ void k_agg(const unsigned char* __restrict__ labT,
                      const _Float16* __restrict__ HlT,
                      _Float16* __restrict__ m16) {
  int t = blockIdx.x * 4 + (threadIdx.x >> 5);  // 0..4095
  int b = t >> 6;                               // 64 waves per batch
  int r = t & 63;
  int jt = r >> 2;                              // 16 j-tiles
  int mg = (r & 3) * 4;                         // first of 4 adjacent m-tiles
  int lane = threadIdx.x & 31;
  const unsigned char* mp = labT + (b * 256 + jt * 16 + (lane & 15)) * 256
                                 + (lane >> 4) * 8;
  v8f acc0 = {}, acc1 = {}, acc2 = {}, acc3 = {};
#pragma unroll
  for (int kb = 0; kb < 8; ++kb) {
    unsigned long long m0 = *(const unsigned long long*)(mp + kb * 32);
    unsigned long long m1 = *(const unsigned long long*)(mp + kb * 32 + 16);
#pragma unroll
    for (int l = 0; l < 4; ++l) {
      v16h a = mask_to_frag(m0, m1, l + 1);
      const _Float16* bb = HlT + ((l * 64 + b) * 256 + mg * 16) * 256 + kb * 32;
      acc0 = wmma16(a, load_b_frag_t(bb + 0 * 16 * 256, 256), acc0);
      acc1 = wmma16(a, load_b_frag_t(bb + 1 * 16 * 256, 256), acc1);
      acc2 = wmma16(a, load_b_frag_t(bb + 2 * 16 * 256, 256), acc2);
      acc3 = wmma16(a, load_b_frag_t(bb + 3 * 16 * 256, 256), acc3);
    }
  }
  _Float16* op = m16 + (b * 256 + jt * 16) * 256 + mg * 16;
  store_c_f16(op + 0 * 16, 256, acc0);
  store_c_f16(op + 1 * 16, 256, acc1);
  store_c_f16(op + 2 * 16, 256, acc2);
  store_c_f16(op + 3 * 16, 256, acc3);
}

// Fused GRU: gi = m@Wih^T+bih, gh = h@Whh^T+bhh, elementwise update.
__global__ void k_gru(const _Float16* __restrict__ m16,
                      const _Float16* __restrict__ h16in,
                      const float* __restrict__ htin,
                      const _Float16* __restrict__ wih,
                      const _Float16* __restrict__ whh,
                      const float* __restrict__ bih, const float* __restrict__ bhh,
                      const float* __restrict__ valid,
                      float* __restrict__ htout, _Float16* __restrict__ h16out) {
  int t = blockIdx.x * 4 + (threadIdx.x >> 5);
  int rt = t >> 4, ct = t & 15;
  const _Float16* ap = m16 + (rt * 16) * 256;
  const _Float16* hp = h16in + (rt * 16) * 256;
  v8f gi0 = {}, gi1 = {}, gi2 = {}, gh0 = {}, gh1 = {}, gh2 = {};
#pragma unroll
  for (int k = 0; k < 8; ++k) {
    v16h a = load_a_frag(ap + k * 32, 256);
    gi0 = wmma16(a, load_b_frag_t(wih + (0 * 256 + ct * 16) * 256 + k * 32, 256), gi0);
    gi1 = wmma16(a, load_b_frag_t(wih + (1 * 256 + ct * 16) * 256 + k * 32, 256), gi1);
    gi2 = wmma16(a, load_b_frag_t(wih + (2 * 256 + ct * 16) * 256 + k * 32, 256), gi2);
  }
#pragma unroll
  for (int k = 0; k < 8; ++k) {
    v16h a = load_a_frag(hp + k * 32, 256);
    gh0 = wmma16(a, load_b_frag_t(whh + (0 * 256 + ct * 16) * 256 + k * 32, 256), gh0);
    gh1 = wmma16(a, load_b_frag_t(whh + (1 * 256 + ct * 16) * 256 + k * 32, 256), gh1);
    gh2 = wmma16(a, load_b_frag_t(whh + (2 * 256 + ct * 16) * 256 + k * 32, 256), gh2);
  }
  int l = threadIdx.x & 31, n = l & 15, half = l >> 4;
#pragma unroll
  for (int r = 0; r < 8; ++r) {
    int row = rt * 16 + r + 8 * half;
    int col = ct * 16 + n;
    float ir = gi0[r] + bih[col],        hr = gh0[r] + bhh[col];
    float iz = gi1[r] + bih[256 + col],  hz = gh1[r] + bhh[256 + col];
    float in_ = gi2[r] + bih[512 + col], hn = gh2[r] + bhh[512 + col];
    float rr = sigmoidf_(ir + hr);
    float zz = sigmoidf_(iz + hz);
    float nc = tanhf(in_ + rr * hn);
    float ho = htin[row * 256 + col];
    float hv = ((1.0f - zz) * nc + zz * ho) * valid[row];
    htout[row * 256 + col] = hv;
    h16out[row * 256 + col] = (_Float16)hv;
  }
}

// Readout: block = 16 nodes, 8 waves. i-net 320->128->256->128->128 (sigmoid,
// then extra sigmoid per reference), j-net 256->128->256->128->128 (sigmoid).
__global__ void k_readout(const float* __restrict__ ht, const float* __restrict__ hin,
                          const float* __restrict__ valid,
                          const _Float16* __restrict__ iW,
                          const _Float16* __restrict__ jW,
                          const float* __restrict__ ib1, const float* __restrict__ ib2,
                          const float* __restrict__ ib3, const float* __restrict__ ib4,
                          const float* __restrict__ jb1, const float* __restrict__ jb2,
                          const float* __restrict__ jb3, const float* __restrict__ jb4,
                          float* __restrict__ partial) {
  __shared__ _Float16 X[16 * 320];
  __shared__ _Float16 A1[16 * 128];
  __shared__ _Float16 A2[16 * 256];
  __shared__ float GATE[16 * 128];
  __shared__ float JOUT[16 * 128];
  int tid = threadIdx.x;
  int nodebase = blockIdx.x * 16;
  for (int idx = tid; idx < 16 * 320; idx += 256) {
    int r = idx / 320, c = idx % 320;
    float v = (c < 256) ? ht[(nodebase + r) * 256 + c]
                        : hin[(nodebase + r) * 64 + (c - 256)];
    X[idx] = (_Float16)v;
  }
  __syncthreads();
  int wave = tid >> 5;
  int l = tid & 31, n = l & 15, half = l >> 4;

  // ---- i network ----
  { // L1: 320 -> 128 relu
    v8f acc = {};
#pragma unroll
    for (int k = 0; k < 10; ++k)
      acc = wmma16(load_a_frag(X + k * 32, 320),
                   load_b_frag_t(iW + (wave * 16) * 320 + k * 32, 320), acc);
    store_relu_f16(A1 + wave * 16, 128, acc, ib1, wave * 16);
  }
  __syncthreads();
  { // L2: 128 -> 256 relu (2 col tiles per wave)
    for (int s = 0; s < 2; ++s) {
      int ct = wave * 2 + s;
      v8f acc = {};
#pragma unroll
      for (int k = 0; k < 4; ++k)
        acc = wmma16(load_a_frag(A1 + k * 32, 128),
                     load_b_frag_t(iW + 40960 + (ct * 16) * 128 + k * 32, 128), acc);
      store_relu_f16(A2 + ct * 16, 256, acc, ib2, ct * 16);
    }
  }
  __syncthreads();
  { // L3: 256 -> 128 relu
    v8f acc = {};
#pragma unroll
    for (int k = 0; k < 8; ++k)
      acc = wmma16(load_a_frag(A2 + k * 32, 256),
                   load_b_frag_t(iW + 73728 + (wave * 16) * 256 + k * 32, 256), acc);
    __syncthreads();
    store_relu_f16(A1 + wave * 16, 128, acc, ib3, wave * 16);
  }
  __syncthreads();
  { // L4: 128 -> 128; gate = sigmoid(sigmoid(z))
    v8f acc = {};
#pragma unroll
    for (int k = 0; k < 4; ++k)
      acc = wmma16(load_a_frag(A1 + k * 32, 128),
                   load_b_frag_t(iW + 106496 + (wave * 16) * 128 + k * 32, 128), acc);
#pragma unroll
    for (int r = 0; r < 8; ++r) {
      float z = acc[r] + ib4[wave * 16 + n];
      GATE[(r + 8 * half) * 128 + wave * 16 + n] = sigmoidf_(sigmoidf_(z));
    }
  }
  __syncthreads();

  // ---- j network (input = X[:, :256]) ----
  { // L1: 256 -> 128 relu
    v8f acc = {};
#pragma unroll
    for (int k = 0; k < 8; ++k)
      acc = wmma16(load_a_frag(X + k * 32, 320),
                   load_b_frag_t(jW + (wave * 16) * 256 + k * 32, 256), acc);
    store_relu_f16(A1 + wave * 16, 128, acc, jb1, wave * 16);
  }
  __syncthreads();
  { // L2: 128 -> 256 relu
    for (int s = 0; s < 2; ++s) {
      int ct = wave * 2 + s;
      v8f acc = {};
#pragma unroll
      for (int k = 0; k < 4; ++k)
        acc = wmma16(load_a_frag(A1 + k * 32, 128),
                     load_b_frag_t(jW + 32768 + (ct * 16) * 128 + k * 32, 128), acc);
      store_relu_f16(A2 + ct * 16, 256, acc, jb2, ct * 16);
    }
  }
  __syncthreads();
  { // L3: 256 -> 128 relu
    v8f acc = {};
#pragma unroll
    for (int k = 0; k < 8; ++k)
      acc = wmma16(load_a_frag(A2 + k * 32, 256),
                   load_b_frag_t(jW + 65536 + (wave * 16) * 256 + k * 32, 256), acc);
    __syncthreads();
    store_relu_f16(A1 + wave * 16, 128, acc, jb3, wave * 16);
  }
  __syncthreads();
  { // L4: 128 -> 128; jout = sigmoid(z)
    v8f acc = {};
#pragma unroll
    for (int k = 0; k < 4; ++k)
      acc = wmma16(load_a_frag(A1 + k * 32, 128),
                   load_b_frag_t(jW + 98304 + (wave * 16) * 128 + k * 32, 128), acc);
#pragma unroll
    for (int r = 0; r < 8; ++r) {
      float z = acc[r] + jb4[wave * 16 + n];
      JOUT[(r + 8 * half) * 128 + wave * 16 + n] = sigmoidf_(z);
    }
  }
  __syncthreads();

  // combine + per-block partial (deterministic; reduced by k_reduce)
  if (tid < 128) {
    float s = 0.0f;
#pragma unroll
    for (int r = 0; r < 16; ++r)
      s += GATE[r * 128 + tid] * JOUT[r * 128 + tid] * valid[nodebase + r];
    partial[blockIdx.x * 128 + tid] = s;
  }
}

__global__ void k_reduce(const float* __restrict__ partial, float* __restrict__ out) {
  int b = blockIdx.x, t = threadIdx.x;
  float s = 0.0f;
#pragma unroll
  for (int k = 0; k < 16; ++k) s += partial[(b * 16 + k) * 128 + t];
  out[b * 128 + t] = s;
}

// ---- host ------------------------------------------------------------------
extern "C" void kernel_launch(void* const* d_in, const int* in_sizes, int n_in,
                              void* d_out, int out_size, void* d_ws, size_t ws_size,
                              hipStream_t stream) {
  (void)in_sizes; (void)n_in; (void)out_size; (void)ws_size;
  const float* g    = (const float*)d_in[0];
  const float* hin  = (const float*)d_in[1];
  const float* e    = (const float*)d_in[2];
  const float* A    = (const float*)d_in[3];
  const float* wih  = (const float*)d_in[4];
  const float* whh  = (const float*)d_in[5];
  const float* bih  = (const float*)d_in[6];
  const float* bhh  = (const float*)d_in[7];
  // pytree leaves: ((W,b) x 4) for i_params then j_params
  const float* iWf[4] = {(const float*)d_in[8],  (const float*)d_in[10],
                         (const float*)d_in[12], (const float*)d_in[14]};
  const float* ibf[4] = {(const float*)d_in[9],  (const float*)d_in[11],
                         (const float*)d_in[13], (const float*)d_in[15]};
  const float* jWf[4] = {(const float*)d_in[16], (const float*)d_in[18],
                         (const float*)d_in[20], (const float*)d_in[22]};
  const float* jbf[4] = {(const float*)d_in[17], (const float*)d_in[19],
                         (const float*)d_in[21], (const float*)d_in[23]};
  float* out = (float*)d_out;

  char* ws = (char*)d_ws;
  float*         ht0   = (float*)(ws + 0);                // 16 MB
  float*         ht1   = (float*)(ws + 16777216);         // 16 MB
  _Float16*      h16a  = (_Float16*)(ws + 33554432);      // 8 MB
  _Float16*      h16b  = (_Float16*)(ws + 41943040);      // 8 MB
  _Float16*      m16   = (_Float16*)(ws + 50331648);      // 8 MB
  _Float16*      HlT   = (_Float16*)(ws + 58720256);      // 32 MB
  unsigned char* labT  = (unsigned char*)(ws + 92274688); // 16 MB
  float*         valid = (float*)(ws + 109051904);        // 64 KB
  _Float16*      A16   = (_Float16*)(ws + 109117440);     // 512 KB
  _Float16*      wih16 = (_Float16*)(ws + 109641728);     // 384 KB
  _Float16*      whh16 = (_Float16*)(ws + 110034944);     // 384 KB
  _Float16*      iW16  = (_Float16*)(ws + 110428160);     // 240 KB
  _Float16*      jW16  = (_Float16*)(ws + 110673920);     // 224 KB
  float*         part  = (float*)(ws + 110903296);        // 512 KB

  auto cvt = [&](const float* s, _Float16* d, int n) {
    k_cvt<<<(n + 255) / 256, 256, 0, stream>>>(s, d, n);
  };
  cvt(A, A16, 4 * 256 * 256);
  cvt(wih, wih16, 768 * 256);
  cvt(whh, whh16, 768 * 256);
  cvt(iWf[0], iW16 + 0,      128 * 320);
  cvt(iWf[1], iW16 + 40960,  256 * 128);
  cvt(iWf[2], iW16 + 73728,  128 * 256);
  cvt(iWf[3], iW16 + 106496, 128 * 128);
  cvt(jWf[0], jW16 + 0,      128 * 256);
  cvt(jWf[1], jW16 + 32768,  256 * 128);
  cvt(jWf[2], jW16 + 65536,  128 * 256);
  cvt(jWf[3], jW16 + 98304,  128 * 128);

  k_lab<<<65536, 256, 0, stream>>>(e, g, labT);  // labels constant across layers
  k_init<<<16384, 256, 0, stream>>>(hin, ht0, h16a);
  k_valid<<<64, 256, 0, stream>>>(hin, valid);

  float* hti = ht0; float* hto = ht1;
  _Float16* h16i = h16a; _Float16* h16o = h16b;
  for (int layer = 0; layer < 3; ++layer) {
    k_msg<<<16384, 128, 0, stream>>>(h16i, A16, HlT);
    k_agg<<<1024, 128, 0, stream>>>(labT, HlT, m16);
    k_gru<<<4096, 128, 0, stream>>>(m16, h16i, hti, wih16, whh16, bih, bhh,
                                    valid, hto, h16o);
    float* tf = hti; hti = hto; hto = tf;
    _Float16* th = h16i; h16i = h16o; h16o = th;
  }

  k_readout<<<1024, 256, 0, stream>>>(hti, hin, valid, iW16, jW16,
                                      ibf[0], ibf[1], ibf[2], ibf[3],
                                      jbf[0], jbf[1], jbf[2], jbf[3], part);
  k_reduce<<<64, 128, 0, stream>>>(part, out);
}
